// Transformer_48558900248827
// MI455X (gfx1250) — compile-verified, matrix operand order checked by reading, not run on hardware
//
#include <hip/hip_runtime.h>
#include <hip/hip_bf16.h>
#include <math.h>

#define B_ 2
#define S_ 1024
#define D_ 1024
#define H_ 16
#define HD_ 64
#define L_ 8
#define V_ 16384
#define FFN_ 2816
#define EPS_ 1e-5f

typedef __attribute__((ext_vector_type(16))) __bf16 v16bf;
typedef __attribute__((ext_vector_type(8)))  float  v8f;
typedef unsigned int u32x4 __attribute__((ext_vector_type(4)));
typedef int          i32x4 __attribute__((ext_vector_type(4)));
typedef int          i32x8 __attribute__((ext_vector_type(8)));

union Frag16 { v16bf v; unsigned int u[8]; };

__device__ __forceinline__ v8f wmma_bf16(v16bf a, v16bf b, v8f c) {
  return __builtin_amdgcn_wmma_f32_16x16x32_bf16(false, a, false, b, (short)0, c,
                                                 false, false);
}

#if __has_builtin(__builtin_amdgcn_tensor_load_to_lds) && \
    __has_builtin(__builtin_amdgcn_s_wait_tensorcnt)
#define HAVE_TDM 1
#else
#define HAVE_TDM 0
#endif

#if HAVE_TDM
// Issue one TDM DMA of a 2D bf16 tile (tile_rows x tile_k elems) from global
// memory (row pitch row_stride_elems) into LDS at lds_byte_off, with hardware
// LDS padding: 2 dwords inserted after every 2^(pad_code) 8-byte units.
// D# layout per CDNA5 ISA. 6-arg builtin:
//   (u32x4 group0, i32x8 group1, i32x4 group2, i32x4 group3, i32x8, i32 cpol)
__device__ __forceinline__ void tdm_load_2d_bf16(const void* gaddr,
                                                 unsigned lds_byte_off,
                                                 unsigned tile_k,
                                                 unsigned tile_rows,
                                                 unsigned row_stride_elems,
                                                 unsigned pad_code) {
  unsigned long long ga = (unsigned long long)(size_t)gaddr;
  u32x4 g0;
  g0[0] = 1u;                                   // count=1, user descriptor
  g0[1] = lds_byte_off;                         // lds_addr (bytes)
  g0[2] = (unsigned)(ga & 0xFFFFFFFFu);         // global_addr[31:0]
  g0[3] = (unsigned)((ga >> 32) & 0x01FFFFFFu)  // global_addr[56:32]
        | 0x80000000u;                          // type=2 ("image")
  i32x8 g1;
  g1[0] = (int)((1u << 16)                      // data_size=1 -> 2 bytes
        | (1u << 20)                            // pad_enable
        | (pad_code << 22)                      // pad_interval
        | (1u << 25));                          // pad_amount code 1 -> 2 dwords
  g1[1] = (int)((tile_k & 0xFFFFu) << 16);      // tensor_dim0[15:0]
  g1[2] = (int)((tile_k >> 16) | ((tile_rows & 0xFFFFu) << 16));  // d0 hi|d1 lo
  g1[3] = (int)((tile_rows >> 16) | (tile_k << 16));   // d1 hi | tile_dim0
  g1[4] = (int)(tile_rows & 0xFFFFu);           // tile_dim1 (tile_dim2 = 0)
  g1[5] = (int)row_stride_elems;                // tensor_dim0_stride[31:0]
  g1[6] = 0;
  g1[7] = 0;
  i32x4 z4 = {0, 0, 0, 0};
  i32x8 z8 = {0, 0, 0, 0, 0, 0, 0, 0};
  __builtin_amdgcn_tensor_load_to_lds(g0, g1, z4, z4, z8, 0);
}
#endif

__device__ __forceinline__ int first_wave() {
  return __builtin_amdgcn_readfirstlane((int)(threadIdx.x >> 5)) == 0;
}

__device__ __forceinline__ void wait_tensor0() {
#if HAVE_TDM
  __builtin_amdgcn_s_wait_tensorcnt(0);
#endif
}

// Gather a 16x32 bf16 fragment from an LDS tile (row-major, ldm elements/row).
// Lane layout per CDNA5 ISA 7.12.2 (16-bit A-matrix 16x32):
//   lanes 0-15: row = lane,     K pairs 0..7  in VGPR0-3, 16..23 in VGPR4-7
//   lanes16-31: row = lane-16,  K pairs 8..15 in VGPR0-3, 24..31 in VGPR4-7
// Same gather works for B fragments when the tile is stored [n][k].
__device__ __forceinline__ void load_frag16(Frag16& f, const __bf16* base,
                                            int ldm, int row16, int lane) {
  const int r  = row16 + (lane & 15);
  const int kh = (lane >> 4) << 2;  // dword offset: 0 or 4 (k = 0 or 8)
  const unsigned int* p = (const unsigned int*)(base + (size_t)r * ldm);
#pragma unroll
  for (int j = 0; j < 4; ++j) f.u[j] = p[kh + j];
#pragma unroll
  for (int j = 0; j < 4; ++j) f.u[4 + j] = p[8 + kh + j];
}

// ---------------------------------------------------------------------------
// GEMM: C[M,N](f32) = A[M,K](bf16) x W[N,K](f32, converted->bf16 in LDS)^T
// optionally C += (residual accumulate in place).
// Block tile 128x128x32, 8 waves, each wave 64x32 (4x2 of 16x16 WMMA tiles).
// Ping-pong LDS buffers: tile t+1 (TDM DMA of A + VALU fp32->bf16 of W) is
// staged while the WMMAs for tile t run.
// ---------------------------------------------------------------------------
#define BM 128
#define BN 128
#define BK 32
#define LDT 36  // 32 bf16 row + 2-dword TDM pad: 72B rows, conflict-free gather

template <bool ADD>
__global__ __launch_bounds__(256) void gemm_bf16(
    const __bf16* __restrict__ A, const float* __restrict__ W,
    float* __restrict__ C, int M, int N, int K) {
  __shared__ __bf16 As[2][BM * LDT];
  __shared__ __bf16 Ws[2][BN * LDT];
  const int tid  = threadIdx.x;
  const int lane = tid & 31;
  const int wave = tid >> 5;
  const int bm = blockIdx.y * BM;
  const int bn = blockIdx.x * BN;
  const int wm = (wave & 1) * 64;
  const int wn = (wave >> 1) * 32;

  v8f acc[4][2];
#pragma unroll
  for (int i = 0; i < 4; ++i)
#pragma unroll
    for (int j = 0; j < 2; ++j)
#pragma unroll
      for (int e = 0; e < 8; ++e) acc[i][j][e] = 0.0f;

  auto stageA = [&](int p, int k0) {
#if HAVE_TDM
    // Async DMA of the 128x32 bf16 A tile into LDS (pad 16dw rows -> 18dw).
    if (first_wave())
      tdm_load_2d_bf16(A + (size_t)bm * K + k0,
                       (unsigned)(size_t)(&As[p][0]), BK, BM, (unsigned)K, 3);
#else
#pragma unroll
    for (int t = 0; t < 4; ++t) {
      int c = tid + 256 * t;
      int row = c >> 3, seg = c & 7;
      *(uint2*)(&As[p][0] + row * LDT + seg * 4) =
          *((const uint2*)(A + (size_t)(bm + row) * K + k0) + seg);
    }
#endif
  };
  auto stageW = [&](int p, int k0) {  // fp32 -> bf16 convert while writing LDS
#pragma unroll
    for (int t = 0; t < 4; ++t) {
      int c = tid + 256 * t;
      int row = c >> 3, seg = c & 7;
      float4 w4 = *((const float4*)(W + (size_t)(bn + row) * K + k0) + seg);
      __bf16* d = &Ws[p][0] + row * LDT + seg * 4;
      d[0] = (__bf16)w4.x; d[1] = (__bf16)w4.y;
      d[2] = (__bf16)w4.z; d[3] = (__bf16)w4.w;
    }
    if (k0 + BK < K) {  // speculative prefetch of the tile after next into GL2
      int row = tid >> 1;
      __builtin_prefetch(W + (size_t)(bn + row) * K + k0 + BK, 0, 0);
    }
  };

  stageA(0, 0);
  stageW(0, 0);
  wait_tensor0();
  __syncthreads();

  const int nk = K / BK;
  for (int t0 = 0; t0 < nk; ++t0) {
    const int p = t0 & 1;
    if (t0 + 1 < nk) {         // stage next tile into the idle buffer
      stageA(p ^ 1, (t0 + 1) * BK);
      stageW(p ^ 1, (t0 + 1) * BK);
    }

    Frag16 af[4], bfg[2];
#pragma unroll
    for (int i = 0; i < 4; ++i)
      load_frag16(af[i], &As[p][0], LDT, wm + i * 16, lane);
#pragma unroll
    for (int j = 0; j < 2; ++j)
      load_frag16(bfg[j], &Ws[p][0], LDT, wn + j * 16, lane);
#pragma unroll
    for (int i = 0; i < 4; ++i)
#pragma unroll
      for (int j = 0; j < 2; ++j)
        acc[i][j] = wmma_bf16(af[i].v, bfg[j].v, acc[i][j]);

    wait_tensor0();            // next-tile DMA done before anyone proceeds
    __syncthreads();           // and everyone done reading buffer p
  }

  // D layout (16x16 f32): VGPR e -> M = e + 8*(lane>>4), N = lane&15.
  const int rsel = (lane >> 4) * 8;
  const int coln = lane & 15;
#pragma unroll
  for (int i = 0; i < 4; ++i)
#pragma unroll
    for (int j = 0; j < 2; ++j) {
      int n = bn + wn + j * 16 + coln;
#pragma unroll
      for (int e = 0; e < 8; ++e) {
        int m = bm + wm + i * 16 + rsel + e;
        size_t idx = (size_t)m * N + n;
        float v = acc[i][j][e];
        C[idx] = ADD ? (C[idx] + v) : v;
      }
    }
}

// ---------------------------------------------------------------------------
// Flash attention, one block per (q-block of 64, head, batch). Causal.
// Q,K: bf16 [B][H][S][HD].  V: pre-transposed bf16 [B][H][HD][S].
// Output: bf16 [B*S][D] (ready as GEMM A operand).
// Q, K and V tiles are TDM-DMA'd (64x64 bf16, hardware-padded 32dw->34dw rows).
// ---------------------------------------------------------------------------
#define LDH  68  // 64 bf16 row + 2-dword TDM pad: 136B rows
#define LDSS 68  // f32 row stride for the score tile

__global__ __launch_bounds__(256) void attn_kernel(
    const __bf16* __restrict__ Q, const __bf16* __restrict__ K,
    const __bf16* __restrict__ Vt_g, __bf16* __restrict__ O) {
  __shared__ __bf16 Qs[64 * LDH];
  __shared__ __bf16 Ks[64 * LDH];
  __shared__ __bf16 Vt[64 * LDH];   // [hd][key]
  __shared__ __bf16 Pb[64 * LDH];   // probabilities (bf16)
  __shared__ float  Ssc[64 * LDSS]; // raw scores
  __shared__ float  mrow[64], lrow[64], srow[64];

  const int tid  = threadIdx.x;
  const int lane = tid & 31;
  const int wave = tid >> 5;
  const int qb = blockIdx.x, hh = blockIdx.y, b = blockIdx.z;
  const int mr = (wave & 3) * 16;       // wave's 16-row slice of the 64x64 tile
  const int nc = (wave >> 2) * 32;      // wave's 32-col slice
  const int rsel = (lane >> 4) * 8;
  const int coln = lane & 15;
  const float inv_sqrt_hd = 0.125f;     // 1/sqrt(64)

  const size_t headBase = ((size_t)b * H_ + hh) * S_ * HD_;
  const __bf16* Qg  = Q + headBase + (size_t)qb * 64 * HD_;
  const __bf16* Vhg = Vt_g + headBase;  // [HD][S] for this (b,h)

#if HAVE_TDM
  if (first_wave())
    tdm_load_2d_bf16(Qg, (unsigned)(size_t)(&Qs[0]), HD_, 64, HD_, 4);
#else
#pragma unroll
  for (int t = 0; t < 4; ++t) {
    int c = tid + 256 * t;
    int row = c >> 4, seg = c & 15;
    *(uint2*)(Qs + row * LDH + seg * 4) =
        *((const uint2*)(Qg + row * HD_) + seg);
  }
#endif
  if (tid < 64) { mrow[tid] = -3.0e38f; lrow[tid] = 0.0f; }

  v8f oacc[2];
#pragma unroll
  for (int j = 0; j < 2; ++j)
#pragma unroll
    for (int e = 0; e < 8; ++e) oacc[j][e] = 0.0f;
  wait_tensor0();
  __syncthreads();

  for (int kb = 0; kb <= qb; ++kb) {
    const __bf16* Kg = K + headBase + (size_t)kb * 64 * HD_;
#if HAVE_TDM
    if (first_wave()) {
      tdm_load_2d_bf16(Kg, (unsigned)(size_t)(&Ks[0]), HD_, 64, HD_, 4);
      tdm_load_2d_bf16(Vhg + (size_t)kb * 64, (unsigned)(size_t)(&Vt[0]),
                       64, 64, S_, 4);  // rows = hd, pitch = S
    }
#else
#pragma unroll
    for (int t = 0; t < 4; ++t) {
      int c = tid + 256 * t;
      int row = c >> 4, seg = c & 15;
      *(uint2*)(Ks + row * LDH + seg * 4) =
          *((const uint2*)(Kg + row * HD_) + seg);
      *(uint2*)(Vt + row * LDH + seg * 4) =
          *((const uint2*)(Vhg + (size_t)row * S_ + kb * 64) + seg);
    }
#endif
    wait_tensor0();
    __syncthreads();

    // S = Q * K^T  (B fragment gathered from Ks stored [key][hd])
    v8f sacc[2];
#pragma unroll
    for (int j = 0; j < 2; ++j)
#pragma unroll
      for (int e = 0; e < 8; ++e) sacc[j][e] = 0.0f;
#pragma unroll
    for (int kc = 0; kc < HD_; kc += 32) {
      Frag16 aq; load_frag16(aq, Qs + kc, LDH, mr, lane);
#pragma unroll
      for (int j = 0; j < 2; ++j) {
        Frag16 bk; load_frag16(bk, Ks + kc, LDH, nc + j * 16, lane);
        sacc[j] = wmma_bf16(aq.v, bk.v, sacc[j]);
      }
    }
#pragma unroll
    for (int j = 0; j < 2; ++j)
#pragma unroll
      for (int e = 0; e < 8; ++e) {
        int ml = mr + rsel + e;
        int nl = nc + j * 16 + coln;
        float s = sacc[j][e] * inv_sqrt_hd;
        if (kb == qb && nl > ml) s = -3.0e38f;  // causal mask, diagonal block
        Ssc[ml * LDSS + nl] = s;
      }
    __syncthreads();

    if (tid < 64) {                     // online softmax, one thread per row
      int r = tid;
      float mo = mrow[r], mx = mo;
      for (int c2 = 0; c2 < 64; ++c2) mx = fmaxf(mx, Ssc[r * LDSS + c2]);
      float sc = __expf(mo - mx);
      float sum = 0.f;
      for (int c2 = 0; c2 < 64; ++c2) {
        float p = __expf(Ssc[r * LDSS + c2] - mx);
        Pb[r * LDH + c2] = (__bf16)p;
        sum += p;
      }
      lrow[r] = lrow[r] * sc + sum;
      mrow[r] = mx;
      srow[r] = sc;
    }
    __syncthreads();

#pragma unroll
    for (int j = 0; j < 2; ++j)
#pragma unroll
      for (int e = 0; e < 8; ++e) oacc[j][e] *= srow[mr + rsel + e];

    // O += P * V  (B fragment gathered from Vt stored [hd][key])
#pragma unroll
    for (int kc = 0; kc < 64; kc += 32) {
      Frag16 ap; load_frag16(ap, Pb + kc, LDH, mr, lane);
#pragma unroll
      for (int j = 0; j < 2; ++j) {
        Frag16 bv; load_frag16(bv, Vt + kc, LDH, nc + j * 16, lane);
        oacc[j] = wmma_bf16(ap.v, bv.v, oacc[j]);
      }
    }
    __syncthreads();
  }

#pragma unroll
  for (int j = 0; j < 2; ++j)
#pragma unroll
    for (int e = 0; e < 8; ++e) {
      int ml = mr + rsel + e;
      int sg = qb * 64 + ml;
      int hd = nc + j * 16 + coln;
      float o = oacc[j][e] / lrow[ml];
      O[((size_t)(b * S_ + sg)) * D_ + hh * HD_ + hd] = (__bf16)o;
    }
}

// ---------------------------------------------------------------------------
// Elementwise kernels
// ---------------------------------------------------------------------------
__global__ __launch_bounds__(256) void rmsnorm_kernel(
    const float* __restrict__ x, const float* __restrict__ w,
    __bf16* __restrict__ out) {
  __shared__ float red[256];
  const int row = blockIdx.x, tid = threadIdx.x;
  const float* xr = x + (size_t)row * D_;
  float ss = 0.f;
  for (int i = tid; i < D_; i += 256) { float v = xr[i]; ss += v * v; }
  red[tid] = ss;
  __syncthreads();
  for (int s = 128; s > 0; s >>= 1) {
    if (tid < s) red[tid] += red[tid + s];
    __syncthreads();
  }
  float inv = rsqrtf(red[0] / (float)D_ + EPS_);
  for (int i = tid; i < D_; i += 256)
    out[(size_t)row * D_ + i] = (__bf16)(xr[i] * inv * w[i]);
}

// qkv f32 [B][S][3D] + freqs [S][32][2]  ->  Q,K (rope) bf16 [B][H][S][HD]
__global__ __launch_bounds__(256) void rope_kernel(
    const float* __restrict__ qkv, const float* __restrict__ fc,
    __bf16* __restrict__ Q, __bf16* __restrict__ K) {
  int t = blockIdx.x * blockDim.x + threadIdx.x;  // over B*S*H*32
  int d2 = t & 31;
  int r = t >> 5;
  int hh = r & (H_ - 1);
  r >>= 4;
  int s = r & (S_ - 1);
  int b = r >> 10;
  size_t base = ((size_t)(b * S_ + s)) * (3 * D_);
  size_t qoff = base + (size_t)hh * HD_ + 2 * d2;
  float q0 = qkv[qoff],      q1 = qkv[qoff + 1];
  float k0 = qkv[qoff + D_], k1 = qkv[qoff + D_ + 1];
  float c  = fc[(s * 32 + d2) * 2 + 0];
  float sn = fc[(s * 32 + d2) * 2 + 1];
  size_t o = ((size_t)(b * H_ + hh) * S_ + s) * HD_ + 2 * d2;
  Q[o]     = (__bf16)(q0 * c - q1 * sn);
  Q[o + 1] = (__bf16)(q1 * c + q0 * sn);
  K[o]     = (__bf16)(k0 * c - k1 * sn);
  K[o + 1] = (__bf16)(k1 * c + k0 * sn);
}

// V slice of qkv f32 [B][S][3D] -> transposed bf16 [B][H][HD][S]
// (LDS-tiled: coalesced reads and coalesced writes)
__global__ __launch_bounds__(256) void vtrans_kernel(
    const float* __restrict__ qkv, __bf16* __restrict__ Vt) {
  __shared__ __bf16 tile[64][65];
  const int sb = blockIdx.x, hh = blockIdx.y, b = blockIdx.z;
  const int tid = threadIdx.x;
#pragma unroll
  for (int t = 0; t < 16; ++t) {
    int e = tid + 256 * t;
    int row = e >> 6, hd = e & 63;  // row = local s
    int s = sb * 64 + row;
    float v = qkv[((size_t)(b * S_ + s)) * (3 * D_) + 2 * D_ + hh * HD_ + hd];
    tile[row][hd] = (__bf16)v;
  }
  __syncthreads();
#pragma unroll
  for (int t = 0; t < 16; ++t) {
    int e = tid + 256 * t;
    int hd = e >> 6, col = e & 63;  // col = local s
    Vt[((size_t)((b * H_ + hh) * HD_ + hd)) * S_ + sb * 64 + col] =
        tile[col][hd];
  }
}

__global__ __launch_bounds__(256) void silumul_kernel(
    const float* __restrict__ a, const float* __restrict__ bb,
    __bf16* __restrict__ g, int n) {
  int t = blockIdx.x * blockDim.x + threadIdx.x;
  if (t < n) {
    float x = a[t];
    float s = x / (1.f + __expf(-x));
    g[t] = (__bf16)(s * bb[t]);
  }
}

// ---------------------------------------------------------------------------
// Host orchestration
// ---------------------------------------------------------------------------
extern "C" void kernel_launch(void* const* d_in, const int* in_sizes, int n_in,
                              void* d_out, int out_size, void* d_ws,
                              size_t ws_size, hipStream_t stream) {
  (void)in_sizes; (void)n_in; (void)out_size; (void)ws_size;
  const float* x    = (const float*)d_in[0];
  const float* fc   = (const float*)d_in[1];
  const float* wqkv = (const float*)d_in[2];
  const float* wo   = (const float*)d_in[3];
  const float* w1   = (const float*)d_in[4];
  const float* w2   = (const float*)d_in[5];
  const float* w3   = (const float*)d_in[6];
  const float* anw  = (const float*)d_in[7];
  const float* fnw  = (const float*)d_in[8];
  const float* finw = (const float*)d_in[9];
  const float* hw   = (const float*)d_in[10];
  float* logits = (float*)d_out;

  char* ws = (char*)d_ws;
  size_t off = 0;
  auto carve = [&](size_t bytes) {
    char* p = ws + off;
    off += (bytes + 255) & ~(size_t)255;
    return p;
  };
  const int M = B_ * S_;
  float*  h   = (float*) carve((size_t)M * D_ * 4);
  __bf16* xn  = (__bf16*)carve((size_t)M * D_ * 2);
  float*  qkv = (float*) carve((size_t)M * 3 * D_ * 4);
  __bf16* Qb  = (__bf16*)carve((size_t)M * D_ * 2);
  __bf16* Kb  = (__bf16*)carve((size_t)M * D_ * 2);
  __bf16* Vb  = (__bf16*)carve((size_t)M * D_ * 2);  // [B][H][HD][S]
  __bf16* ao  = (__bf16*)carve((size_t)M * D_ * 2);
  float*  f1  = (float*) carve((size_t)M * FFN_ * 4);
  float*  f3  = (float*) carve((size_t)M * FFN_ * 4);
  __bf16* g   = (__bf16*)carve((size_t)M * FFN_ * 2);

  (void)hipMemcpyAsync(h, x, (size_t)M * D_ * 4, hipMemcpyDeviceToDevice,
                       stream);

  dim3 blk(256);
  for (int l = 0; l < L_; ++l) {
    rmsnorm_kernel<<<M, blk, 0, stream>>>(h, anw + (size_t)l * D_, xn);
    gemm_bf16<false><<<dim3(3 * D_ / BN, M / BM), blk, 0, stream>>>(
        xn, wqkv + (size_t)l * 3 * D_ * D_, qkv, M, 3 * D_, D_);
    rope_kernel<<<(M * H_ * 32) / 256, blk, 0, stream>>>(qkv, fc, Qb, Kb);
    vtrans_kernel<<<dim3(S_ / 64, H_, B_), blk, 0, stream>>>(qkv, Vb);
    attn_kernel<<<dim3(S_ / 64, H_, B_), blk, 0, stream>>>(Qb, Kb, Vb, ao);
    gemm_bf16<true><<<dim3(D_ / BN, M / BM), blk, 0, stream>>>(
        ao, wo + (size_t)l * D_ * D_, h, M, D_, D_);
    rmsnorm_kernel<<<M, blk, 0, stream>>>(h, fnw + (size_t)l * D_, xn);
    gemm_bf16<false><<<dim3(FFN_ / BN, M / BM), blk, 0, stream>>>(
        xn, w1 + (size_t)l * FFN_ * D_, f1, M, FFN_, D_);
    gemm_bf16<false><<<dim3(FFN_ / BN, M / BM), blk, 0, stream>>>(
        xn, w3 + (size_t)l * FFN_ * D_, f3, M, FFN_, D_);
    silumul_kernel<<<(M * FFN_) / 256, blk, 0, stream>>>(f1, f3, g, M * FFN_);
    gemm_bf16<true><<<dim3(D_ / BN, M / BM), blk, 0, stream>>>(
        g, w2 + (size_t)l * D_ * FFN_, h, M, D_, FFN_);
  }
  rmsnorm_kernel<<<M, blk, 0, stream>>>(h, finw, xn);
  gemm_bf16<false><<<dim3(V_ / BN, M / BM), blk, 0, stream>>>(
      xn, hw, logits, M, V_, D_);
}
